// RelTransformerBlock_64914135712025
// MI455X (gfx1250) — compile-verified, matrix operand order checked by reading, not run on hardware
//
#include <hip/hip_runtime.h>
#include <hip/hip_bf16.h>

// ---------------------------------------------------------------------------
// Types / helpers
// ---------------------------------------------------------------------------
typedef __bf16 v16bf __attribute__((ext_vector_type(16)));
typedef float  v8f   __attribute__((ext_vector_type(8)));
typedef unsigned int u32x4 __attribute__((ext_vector_type(4)));

union BF16x16 { v16bf v; __bf16 h[16]; u32x4 q[2]; };
union BF16x32 { u32x4 q[4]; __bf16 h[32]; };
union F8      { v8f   v; float  f[8];  };

#define NEG_MIN_F (-3.402823466e38f)

static __device__ inline __bf16 f2bf(float f) {
  unsigned int u = __builtin_bit_cast(unsigned int, f);
  unsigned int r = u + 0x7fffu + ((u >> 16) & 1u);       // round-to-nearest-even
  unsigned short s = (unsigned short)(r >> 16);
  return __builtin_bit_cast(__bf16, s);
}
static __device__ inline float bf2f(__bf16 h) {
  unsigned int u = ((unsigned int)__builtin_bit_cast(unsigned short, h)) << 16;
  return __builtin_bit_cast(float, u);
}
static __device__ inline v8f v8f_zero() {
  v8f z = {0.f,0.f,0.f,0.f,0.f,0.f,0.f,0.f};
  return z;
}
static __device__ inline v8f wmma_bf16(v16bf a, v16bf b, v8f c) {
  // v_wmma_f32_16x16x32_bf16  (neg_a, A, neg_b, B, c_mod, C, reuse_a, reuse_b)
  return __builtin_amdgcn_wmma_f32_16x16x32_bf16(false, a, false, b, (short)0, c, false, false);
}

// Fragment from a row-major [rows][K] matrix: rows r0+lane%16,
// K = k0 + kmap(e, lane>=16) per the CDNA5 16-bit A-matrix layout.
// Requires: base 16B-aligned, (ld*2)%16==0, (k0*2)%16==0  -> 2x ds_load_b128.
static __device__ inline v16bf load_frag_nk(const __bf16* base, int ld, int r0, int k0, int lane) {
  const __bf16* p = base + (size_t)(r0 + (lane & 15)) * ld + k0 + ((lane >> 4) << 3);
  BF16x16 f;
  f.q[0] = *(const u32x4*)p;
  f.q[1] = *(const u32x4*)(p + 16);
  return f.v;
}

// ---------------------------------------------------------------------------
// f32 -> bf16 conversion
// ---------------------------------------------------------------------------
__global__ void cvt_kernel(const float* __restrict__ in, __bf16* __restrict__ out, int n) {
  int i = blockIdx.x * blockDim.x + threadIdx.x;
  int stride = gridDim.x * blockDim.x;
  for (; i < n; i += stride) out[i] = f2bf(in[i]);
}

// ---------------------------------------------------------------------------
// bf16 WMMA GEMM: C[M,N] = A[M,K] @ B[K,N]  (+bias, +relu), f32 and/or bf16 out
// Block 256 thr (8 waves), tile 128x128, K-step 32. M%128==N%128==K%32==0.
// B is staged transposed into LDS so all fragment loads are ds_load_b128.
// ---------------------------------------------------------------------------
__global__ __launch_bounds__(256) void gemm_kernel(
    const __bf16* __restrict__ A, const __bf16* __restrict__ B,
    const float* __restrict__ bias, float* __restrict__ Cf, __bf16* __restrict__ Cb,
    int M, int N, int K, int relu)
{
  __shared__ __attribute__((aligned(16))) __bf16 As[128][32];
  __shared__ __attribute__((aligned(16))) __bf16 BsT[128][32];   // [n][k]
  (void)M;
  const int tid = threadIdx.x, lane = tid & 31, wv = tid >> 5;
  const int wm = wv >> 1, wn = wv & 1;               // 4x2 wave grid
  const int hi = lane >> 4, ln = lane & 15;
  const int bm = blockIdx.y * 128, bn = blockIdx.x * 128;
  const int ar = tid >> 1, ac = (tid & 1) * 16;      // A: 2 thr/row of 32
  const int br = tid >> 3, bc = (tid & 7) * 16;      // B: 8 thr/row of 128

  F8 acc[2][4];
#pragma unroll
  for (int ms = 0; ms < 2; ++ms)
#pragma unroll
    for (int ns = 0; ns < 4; ++ns) acc[ms][ns].v = v8f_zero();

  for (int k0 = 0; k0 < K; k0 += 32) {
    __syncthreads();
    {
      const u32x4* g = (const u32x4*)(A + (size_t)(bm + ar) * K + k0 + ac);
      u32x4* sp = (u32x4*)&As[ar][ac];
      sp[0] = g[0]; sp[1] = g[1];
    }
    {
      BF16x16 bb;
      const u32x4* g = (const u32x4*)(B + (size_t)(k0 + br) * N + bn + bc);
      bb.q[0] = g[0]; bb.q[1] = g[1];
#pragma unroll
      for (int e = 0; e < 16; ++e) BsT[bc + e][br] = bb.h[e];
    }
    if (k0 + 32 < K) {  // global_prefetch_b8 of next K tile
      __builtin_prefetch(A + (size_t)(bm + ar) * K + (k0 + 32) + ac, 0, 0);
      __builtin_prefetch(B + (size_t)(k0 + 32 + br) * N + bn + bc, 0, 0);
    }
    __syncthreads();

    v16bf a[2];
#pragma unroll
    for (int ms = 0; ms < 2; ++ms)
      a[ms] = load_frag_nk(&As[0][0], 32, wm * 32 + 16 * ms, 0, lane);
#pragma unroll
    for (int ns = 0; ns < 4; ++ns) {
      v16bf bf = load_frag_nk(&BsT[0][0], 32, wn * 64 + 16 * ns, 0, lane);
#pragma unroll
      for (int ms = 0; ms < 2; ++ms)
        acc[ms][ns].v = wmma_bf16(a[ms], bf, acc[ms][ns].v);
    }
  }

#pragma unroll
  for (int ms = 0; ms < 2; ++ms)
#pragma unroll
    for (int ns = 0; ns < 4; ++ns)
#pragma unroll
      for (int v = 0; v < 8; ++v) {
        int row = bm + wm * 32 + 16 * ms + v + 8 * hi;
        int col = bn + wn * 64 + 16 * ns + ln;
        float x = acc[ms][ns].f[v];
        if (bias) x += bias[col];
        if (relu) x = fmaxf(x, 0.f);
        size_t o = (size_t)row * N + col;
        if (Cf) Cf[o] = x;
        if (Cb) Cb[o] = f2bf(x);
      }
}

// ---------------------------------------------------------------------------
// Fused TXL relative attention (flash-style), one (head,batch,64-query-tile)
// per block of 128 threads (4 waves; wave = 16 query rows).
//   S = (Qw@K^T + gather(Qr@PrelBand^T)) * scale ; causal mask ; online softmax
//   O += P@V ; avec = O / l   (bf16 out, [q*2+b][h*64+d])
// heads: bf16 [4096][3072] = {q|k|v}; rk: bf16 [2048][1024]
// ---------------------------------------------------------------------------
#define L_SEQ 2048
#define QT 64
#define DH 64

__global__ __launch_bounds__(128) void attn_kernel(
    const __bf16* __restrict__ heads, const __bf16* __restrict__ rk,
    const float* __restrict__ rwb, const float* __restrict__ rrb,
    __bf16* __restrict__ avec)
{
  __shared__ __attribute__((aligned(16))) __bf16 Qw[QT][DH];
  __shared__ __attribute__((aligned(16))) __bf16 Qr[QT][DH];
  __shared__ __attribute__((aligned(16))) __bf16 Kt[QT][DH];    // [j][d]
  __shared__ __attribute__((aligned(16))) __bf16 VtT[DH][QT];   // [d][j]
  __shared__ __attribute__((aligned(16))) __bf16 Pband[128][DH];
  __shared__ __attribute__((aligned(16))) __bf16 Pp[QT][QT];    // P staging
  __shared__ float Gs[4][16][80];                               // per-wave band

  const int tid = threadIdx.x, lane = tid & 31, wv = tid >> 5;
  const int it = blockIdx.x, h = blockIdx.y, b = blockIdx.z;
  const int i0 = it * QT;
  const int hi = lane >> 4, ln = lane & 15;

  // Stage Q with r_w_bias / r_r_bias added (bf16 re-round); vectorized loads.
  {
    int ql = tid >> 1, d0 = (tid & 1) * 32;
    BF16x32 qq, qw2, qr2;
    const u32x4* g = (const u32x4*)(heads + (size_t)((i0 + ql) * 2 + b) * 3072 + h * DH + d0);
    qq.q[0] = g[0]; qq.q[1] = g[1]; qq.q[2] = g[2]; qq.q[3] = g[3];
#pragma unroll
    for (int e = 0; e < 32; ++e) {
      float f = bf2f(qq.h[e]);
      qw2.h[e] = f2bf(f + rwb[h * DH + d0 + e]);
      qr2.h[e] = f2bf(f + rrb[h * DH + d0 + e]);
    }
    u32x4* sw = (u32x4*)&Qw[ql][d0];
    u32x4* sr = (u32x4*)&Qr[ql][d0];
#pragma unroll
    for (int e = 0; e < 4; ++e) { sw[e] = qw2.q[e]; sr[e] = qr2.q[e]; }
  }
  __syncthreads();

  v16bf aQw[2], aQr[2];
#pragma unroll
  for (int kk = 0; kk < 2; ++kk) {
    aQw[kk] = load_frag_nk(&Qw[0][0], DH, 16 * wv, 32 * kk, lane);
    aQr[kk] = load_frag_nk(&Qr[0][0], DH, 16 * wv, 32 * kk, lane);
  }

  float mrow[8], lrow[8];
  F8 o[4];
#pragma unroll
  for (int v = 0; v < 8; ++v) { mrow[v] = NEG_MIN_F; lrow[v] = 0.f; }
#pragma unroll
  for (int ds = 0; ds < 4; ++ds) o[ds].v = v8f_zero();

  for (int jt = 0; jt <= it; ++jt) {
    const int j0 = jt * QT;
    __syncthreads();
    // Stage K tile (b128 copies) and V^T tile (vector load + transpose scatter)
    {
      int jl = tid >> 1, d0 = (tid & 1) * 32;
      size_t row = (size_t)((j0 + jl) * 2 + b) * 3072;
      const u32x4* gk = (const u32x4*)(heads + row + 1024 + h * DH + d0);
      u32x4* sk = (u32x4*)&Kt[jl][d0];
      sk[0] = gk[0]; sk[1] = gk[1]; sk[2] = gk[2]; sk[3] = gk[3];
      BF16x32 vvv;
      const u32x4* gv = (const u32x4*)(heads + row + 2048 + h * DH + d0);
      vvv.q[0] = gv[0]; vvv.q[1] = gv[1]; vvv.q[2] = gv[2]; vvv.q[3] = gv[3];
#pragma unroll
      for (int e = 0; e < 32; ++e) VtT[d0 + e][jl] = vvv.h[e];
    }
    // Stage Prel band: Pband[u] = Rk[L-1-(i0-j0-63+u)] (clamped; OOB is masked)
    {
      int u = tid;
      int rr = L_SEQ + 62 - i0 + j0 - u;
      rr = rr < 0 ? 0 : (rr > L_SEQ - 1 ? L_SEQ - 1 : rr);
      const u32x4* gp = (const u32x4*)(rk + (size_t)rr * 1024 + h * DH);
      u32x4* sp = (u32x4*)&Pband[u][0];
#pragma unroll
      for (int e = 0; e < 8; ++e) sp[e] = gp[e];
    }
    __syncthreads();

    // AC = Qw @ K^T
    F8 s[4];
#pragma unroll
    for (int ns = 0; ns < 4; ++ns) s[ns].v = v8f_zero();
#pragma unroll
    for (int kk = 0; kk < 2; ++kk)
#pragma unroll
      for (int ns = 0; ns < 4; ++ns) {
        v16bf bk = load_frag_nk(&Kt[0][0], DH, 16 * ns, 32 * kk, lane);
        s[ns].v = wmma_bf16(aQw[kk], bk, s[ns].v);
      }

    // G' = Qr @ PrelBand^T over this wave's 80-wide sub-band -> LDS
#pragma unroll
    for (int nsg = 0; nsg < 5; ++nsg) {
      F8 g; g.v = v8f_zero();
#pragma unroll
      for (int kk = 0; kk < 2; ++kk) {
        v16bf bp = load_frag_nk(&Pband[0][0], DH, 16 * wv + 16 * nsg, 32 * kk, lane);
        g.v = wmma_bf16(aQr[kk], bp, g.v);
      }
#pragma unroll
      for (int v = 0; v < 8; ++v)
        Gs[wv][v + 8 * hi][16 * nsg + ln] = g.f[v];
    }

    // S = (AC + shift(BD)) * scale ; causal mask
#pragma unroll
    for (int ns = 0; ns < 4; ++ns)
#pragma unroll
      for (int v = 0; v < 8; ++v) {
        int mr = v + 8 * hi;
        float bd = Gs[wv][mr][mr + 63 - (16 * ns + ln)];
        float sv = (s[ns].f[v] + bd) * 0.125f;
        int iabs = i0 + 16 * wv + mr;
        int jabs = j0 + 16 * ns + ln;
        s[ns].f[v] = (jabs > iabs) ? NEG_MIN_F : sv;
      }

    // Online softmax update
    float corr[8], rsum[8];
#pragma unroll
    for (int v = 0; v < 8; ++v) {
      float r = s[0].f[v];
#pragma unroll
      for (int ns = 1; ns < 4; ++ns) r = fmaxf(r, s[ns].f[v]);
#pragma unroll
      for (int off2 = 8; off2 >= 1; off2 >>= 1) r = fmaxf(r, __shfl_xor(r, off2, 32));
      float mn = fmaxf(mrow[v], r);
      corr[v] = __expf(mrow[v] - mn);
      mrow[v] = mn;
      rsum[v] = 0.f;
    }
#pragma unroll
    for (int ns = 0; ns < 4; ++ns)
#pragma unroll
      for (int v = 0; v < 8; ++v) {
        float p = __expf(s[ns].f[v] - mrow[v]);
        s[ns].f[v] = p;
        rsum[v] += p;
      }
#pragma unroll
    for (int v = 0; v < 8; ++v) {
      float r = rsum[v];
#pragma unroll
      for (int off2 = 8; off2 >= 1; off2 >>= 1) r += __shfl_xor(r, off2, 32);
      lrow[v] = lrow[v] * corr[v] + r;
    }
#pragma unroll
    for (int ds = 0; ds < 4; ++ds)
#pragma unroll
      for (int v = 0; v < 8; ++v) o[ds].f[v] *= corr[v];

    // Stage P (wave-private rows), then O += P @ V
#pragma unroll
    for (int ns = 0; ns < 4; ++ns)
#pragma unroll
      for (int v = 0; v < 8; ++v)
        Pp[16 * wv + v + 8 * hi][16 * ns + ln] = f2bf(s[ns].f[v]);
#pragma unroll
    for (int kk = 0; kk < 2; ++kk) {
      v16bf ap = load_frag_nk(&Pp[0][0], QT, 16 * wv, 32 * kk, lane);
#pragma unroll
      for (int ds = 0; ds < 4; ++ds) {
        v16bf bv = load_frag_nk(&VtT[0][0], QT, 16 * ds, 32 * kk, lane);
        o[ds].v = wmma_bf16(ap, bv, o[ds].v);
      }
    }
  }

  // Epilogue: O / l -> avec (bf16)
#pragma unroll
  for (int v = 0; v < 8; ++v) lrow[v] = 1.f / lrow[v];
#pragma unroll
  for (int ds = 0; ds < 4; ++ds)
#pragma unroll
    for (int v = 0; v < 8; ++v) {
      int mr = v + 8 * hi;
      int iabs = i0 + 16 * wv + mr;
      avec[(size_t)(iabs * 2 + b) * 1024 + h * DH + 16 * ds + ln] =
          f2bf(o[ds].f[v] * lrow[v]);
    }
}

// ---------------------------------------------------------------------------
// Residual + LayerNorm (N == 1024): out = LN(x1 + x2) * g + b
// ---------------------------------------------------------------------------
__global__ __launch_bounds__(256) void ln_kernel(
    const float* __restrict__ x1, const float* __restrict__ x2,
    const float* __restrict__ gam, const float* __restrict__ bet,
    float* __restrict__ outf, __bf16* __restrict__ outb, int N)
{
  __shared__ float sA[256], sB[256];
  const int row = blockIdx.x, tid = threadIdx.x;
  float v[4];
  float s = 0.f, s2 = 0.f;
#pragma unroll
  for (int k = 0; k < 4; ++k) {
    int c = tid + k * 256;
    float x = x1[(size_t)row * N + c] + x2[(size_t)row * N + c];
    v[k] = x; s += x; s2 += x * x;
  }
  sA[tid] = s; sB[tid] = s2;
  __syncthreads();
  for (int off2 = 128; off2 > 0; off2 >>= 1) {
    if (tid < off2) { sA[tid] += sA[tid + off2]; sB[tid] += sB[tid + off2]; }
    __syncthreads();
  }
  float mean = sA[0] / N;
  float var  = sB[0] / N - mean * mean;
  float rs = rsqrtf(var + 1e-5f);
#pragma unroll
  for (int k = 0; k < 4; ++k) {
    int c = tid + k * 256;
    float y = (v[k] - mean) * rs * gam[c] + bet[c];
    outf[(size_t)row * N + c] = y;
    if (outb) outb[(size_t)row * N + c] = f2bf(y);
  }
}

// ---------------------------------------------------------------------------
// Orchestration
// ---------------------------------------------------------------------------
extern "C" void kernel_launch(void* const* d_in, const int* in_sizes, int n_in,
                              void* d_out, int out_size, void* d_ws, size_t ws_size,
                              hipStream_t stream) {
  (void)in_sizes; (void)n_in; (void)out_size; (void)ws_size;
  const float* w      = (const float*)d_in[0];   // [2048,2,1024]
  const float* r      = (const float*)d_in[1];   // [2048,1024]
  const float* rwb    = (const float*)d_in[2];   // [16,64]
  const float* rrb    = (const float*)d_in[3];   // [16,64]
  /* d_in[4] attn_mask: causal, computed analytically */
  const float* qkv_w  = (const float*)d_in[5];   // [1024,3072]
  const float* rnet_w = (const float*)d_in[6];   // [1024,1024]
  const float* o_w    = (const float*)d_in[7];   // [1024,1024]
  const float* ln1_g  = (const float*)d_in[8];
  const float* ln1_b  = (const float*)d_in[9];
  const float* ffn_w1 = (const float*)d_in[10];  // [1024,4096]
  const float* ffn_b1 = (const float*)d_in[11];  // [4096]
  const float* ffn_w2 = (const float*)d_in[12];  // [4096,1024]
  const float* ffn_b2 = (const float*)d_in[13];  // [1024]
  const float* ln2_g  = (const float*)d_in[14];
  const float* ln2_b  = (const float*)d_in[15];
  float* out = (float*)d_out;

  const int Mtok = 4096;   // qlen * bsz
  const int DM = 1024, QKV = 3072, DI = 4096, L = 2048;

  // bump allocator over workspace
  char* ws = (char*)d_ws;
  size_t off = 0;
  auto alloc = [&](size_t bytes) -> char* {
    char* p = ws + off;
    off += (bytes + 255) & ~(size_t)255;
    return p;
  };
  __bf16* wbf    = (__bf16*)alloc((size_t)Mtok * DM * 2);
  __bf16* rbf    = (__bf16*)alloc((size_t)L * DM * 2);
  __bf16* qkvwb  = (__bf16*)alloc((size_t)DM * QKV * 2);
  __bf16* rnetwb = (__bf16*)alloc((size_t)DM * DM * 2);
  __bf16* owb    = (__bf16*)alloc((size_t)DM * DM * 2);
  __bf16* w1b    = (__bf16*)alloc((size_t)DM * DI * 2);
  __bf16* w2b    = (__bf16*)alloc((size_t)DI * DM * 2);
  __bf16* heads  = (__bf16*)alloc((size_t)Mtok * QKV * 2);
  __bf16* rk     = (__bf16*)alloc((size_t)L * DM * 2);
  __bf16* avec   = (__bf16*)alloc((size_t)Mtok * DM * 2);
  float*  aout   = (float*)alloc((size_t)Mtok * DM * 4);
  float*  xf     = (float*)alloc((size_t)Mtok * DM * 4);
  __bf16* xb     = (__bf16*)alloc((size_t)Mtok * DM * 2);
  __bf16* hb     = (__bf16*)alloc((size_t)Mtok * DI * 2);
  float*  core   = (float*)alloc((size_t)Mtok * DM * 4);

  auto cvt = [&](const float* src, __bf16* dst, int n) {
    int blocks = (n + 255) / 256; if (blocks > 16384) blocks = 16384;
    cvt_kernel<<<blocks, 256, 0, stream>>>(src, dst, n);
  };
  cvt(w,      wbf,    Mtok * DM);
  cvt(r,      rbf,    L * DM);
  cvt(qkv_w,  qkvwb,  DM * QKV);
  cvt(rnet_w, rnetwb, DM * DM);
  cvt(o_w,    owb,    DM * DM);
  cvt(ffn_w1, w1b,    DM * DI);
  cvt(ffn_w2, w2b,    DI * DM);

  // 1) QKV projection: heads = w @ qkv_w  (bf16 out)
  gemm_kernel<<<dim3(QKV / 128, Mtok / 128), 256, 0, stream>>>(
      wbf, qkvwb, nullptr, nullptr, heads, Mtok, QKV, DM, 0);
  // 2) rel-pos projection: rk = r @ r_net_w  (bf16 out)
  gemm_kernel<<<dim3(DM / 128, L / 128), 256, 0, stream>>>(
      rbf, rnetwb, nullptr, nullptr, rk, L, DM, DM, 0);
  // 3) fused relative attention -> avec (bf16)
  attn_kernel<<<dim3(L / 64, 16, 2), 128, 0, stream>>>(heads, rk, rwb, rrb, avec);
  // 4) output projection: aout = avec @ o_w  (f32 out)
  gemm_kernel<<<dim3(DM / 128, Mtok / 128), 256, 0, stream>>>(
      avec, owb, nullptr, aout, nullptr, Mtok, DM, DM, 0);
  // 5) x = LN(w + aout)  (f32 + bf16)
  ln_kernel<<<Mtok, 256, 0, stream>>>(w, aout, ln1_g, ln1_b, xf, xb, DM);
  // 6) hb = relu(x @ ffn_w1 + b1)  (bf16)
  gemm_kernel<<<dim3(DI / 128, Mtok / 128), 256, 0, stream>>>(
      xb, w1b, ffn_b1, nullptr, hb, Mtok, DI, DM, 1);
  // 7) core = hb @ ffn_w2 + b2  (f32)
  gemm_kernel<<<dim3(DM / 128, Mtok / 128), 256, 0, stream>>>(
      hb, w2b, ffn_b2, core, nullptr, Mtok, DM, DI, 0);
  // 8) out = LN(x + core)
  ln_kernel<<<Mtok, 256, 0, stream>>>(xf, core, ln2_g, ln2_b, out, nullptr, DM);
}
